// TopkRouter_16320875725187
// MI455X (gfx1250) — compile-verified, matrix operand order checked by reading, not run on hardware
//
#include <hip/hip_runtime.h>
#include <hip/hip_bf16.h>

// ---------------------------------------------------------------------------
// Fused DeepSeek-V3-style top-k router for MI455X (gfx1250, wave32, WMMA).
//   Prologue: pack W [256,2048] f32 -> bf16 WMMA B-fragments in workspace
//             (layout [E/16][H/32][lane][16] = per-lane operand layout).
//   Main:     logits = h @ W^T via v_wmma_f32_16x16x32_bf16 (f32 accum)
//             -> sigmoid -> group top-2 -> top-4 groups -> top-8 experts
//             -> normalized weights * 2.5.
// One block (4 waves / 128 threads) owns 16 tokens x 256 experts.
// ---------------------------------------------------------------------------

typedef __attribute__((ext_vector_type(16))) __bf16 v16bf;
typedef __attribute__((ext_vector_type(8)))  float  v8f;
typedef __attribute__((ext_vector_type(4)))  float  f32x4;

#define H_DIM   2048
#define E_DIM   256
#define KBLK    (H_DIM / 32)   // 64 K-blocks of 32
#define N_GRP   8
#define GRP_SZ  32             // E_DIM / N_GRP
#define TOPK_G  4
#define TOPK_E  8
#define SCALE_R 2.5f

// Convert one lane's 16 bf16 fragment elements from f32 memory.
// 16-bit operand layout (wave32): elements 0..7 <- mem[p+0..7],
// elements 8..15 <- mem[p+16..23]; caller pre-offsets p by (lane>>4)*8.
__device__ __forceinline__ v16bf cvt_frag(const float* __restrict__ p) {
    const f32x4* a = reinterpret_cast<const f32x4*>(p);
    f32x4 x0 = a[0];
    f32x4 x1 = a[1];
    const f32x4* b = reinterpret_cast<const f32x4*>(p + 16);
    f32x4 y0 = b[0];
    f32x4 y1 = b[1];
    v16bf f;
    f[0]  = (__bf16)x0.x; f[1]  = (__bf16)x0.y; f[2]  = (__bf16)x0.z; f[3]  = (__bf16)x0.w;
    f[4]  = (__bf16)x1.x; f[5]  = (__bf16)x1.y; f[6]  = (__bf16)x1.z; f[7]  = (__bf16)x1.w;
    f[8]  = (__bf16)y0.x; f[9]  = (__bf16)y0.y; f[10] = (__bf16)y0.z; f[11] = (__bf16)y0.w;
    f[12] = (__bf16)y1.x; f[13] = (__bf16)y1.y; f[14] = (__bf16)y1.z; f[15] = (__bf16)y1.w;
    return f;
}

// Prologue: pack weights into pre-swizzled bf16 B-fragments.
// fB[((nt*KBLK + kb)*32 + lane)*16 + e]  (one v16bf per (tile,kblock,lane))
__global__ __launch_bounds__(256)
void pack_weights_kernel(const float* __restrict__ wt, __bf16* __restrict__ fB) {
    const int idx  = blockIdx.x * 256 + threadIdx.x;  // (E/16)*KBLK*32 = 32768
    const int lane = idx & 31;
    const int kb   = (idx >> 5) & (KBLK - 1);
    const int nt   = idx >> 11;
    const int half = lane >> 4;
    const int l16  = lane & 15;
    const float* src = wt + (size_t)(nt * 16 + l16) * H_DIM + kb * 32 + half * 8;
    v16bf f = cvt_frag(src);
    *reinterpret_cast<v16bf*>(fB + (size_t)idx * 16) = f;
}

template <bool PACKED>
__global__ __launch_bounds__(128)
void topk_router_kernel(const float*  __restrict__ hs,    // [T, H]
                        const float*  __restrict__ wt,    // [E, H] (fallback)
                        const __bf16* __restrict__ fB,    // packed B fragments
                        const float*  __restrict__ bias,  // [E]
                        float* __restrict__ out,          // [T*8 idx][T*8 w]
                        int T) {
    __shared__ float s_scores[16][E_DIM];   // sigmoid(logits), uncorrected
    __shared__ float s_bias[E_DIM];
    __shared__ float s_gscore[16][N_GRP];

    const int tid   = threadIdx.x;
    const int wave  = tid >> 5;        // 0..3
    const int lane  = tid & 31;        // wave32 lane
    const int half  = lane >> 4;       // 0 or 1
    const int l16   = lane & 15;
    const int t0    = blockIdx.x * 16; // first token of this block

    // stage bias into LDS
    s_bias[tid]       = bias[tid];
    s_bias[tid + 128] = bias[tid + 128];

    // -------------------- GEMM phase: 16 tokens x 256 experts ---------------
    const int nbase = wave * 64;
    const int koff  = half * 8;        // lane-half K offset within 32-step

    const float* aP = hs + (size_t)(t0 + l16) * H_DIM + koff;

    // packed-B per-tile base pointers (stride per k-block = 32 lanes * 16)
    const __bf16* fB0 = fB + ((size_t)(((nbase >> 4) + 0) * KBLK) * 32 + lane) * 16;
    const __bf16* fB1 = fB + ((size_t)(((nbase >> 4) + 1) * KBLK) * 32 + lane) * 16;
    const __bf16* fB2 = fB + ((size_t)(((nbase >> 4) + 2) * KBLK) * 32 + lane) * 16;
    const __bf16* fB3 = fB + ((size_t)(((nbase >> 4) + 3) * KBLK) * 32 + lane) * 16;

    // fallback direct-f32 B row pointers
    const float* bP0 = wt + (size_t)(nbase +  0 + l16) * H_DIM + koff;
    const float* bP1 = wt + (size_t)(nbase + 16 + l16) * H_DIM + koff;
    const float* bP2 = wt + (size_t)(nbase + 32 + l16) * H_DIM + koff;
    const float* bP3 = wt + (size_t)(nbase + 48 + l16) * H_DIM + koff;

    v8f acc0 = {}, acc1 = {}, acc2 = {}, acc3 = {};

    for (int kb = 0; kb < H_DIM; kb += 32) {
        v16bf af = cvt_frag(aP + kb);
        v16bf b0, b1, b2, b3;
        if (PACKED) {
            const size_t off = (size_t)(kb >> 5) * (32 * 16);
            b0 = *reinterpret_cast<const v16bf*>(fB0 + off);
            b1 = *reinterpret_cast<const v16bf*>(fB1 + off);
            b2 = *reinterpret_cast<const v16bf*>(fB2 + off);
            b3 = *reinterpret_cast<const v16bf*>(fB3 + off);
        } else {
            b0 = cvt_frag(bP0 + kb);
            b1 = cvt_frag(bP1 + kb);
            b2 = cvt_frag(bP2 + kb);
            b3 = cvt_frag(bP3 + kb);
        }
        acc0 = __builtin_amdgcn_wmma_f32_16x16x32_bf16(false, af, false, b0,
                                                       (short)0, acc0, false, false);
        acc1 = __builtin_amdgcn_wmma_f32_16x16x32_bf16(false, af, false, b1,
                                                       (short)0, acc1, false, false);
        acc2 = __builtin_amdgcn_wmma_f32_16x16x32_bf16(false, af, false, b2,
                                                       (short)0, acc2, false, false);
        acc3 = __builtin_amdgcn_wmma_f32_16x16x32_bf16(false, af, false, b3,
                                                       (short)0, acc3, false, false);
    }

    // sigmoid + stash scores in LDS.
    // C/D layout: VGPR r -> token row (half? 8+r : r); lane%16 -> expert col.
    const int rowbase = half * 8;
#pragma unroll
    for (int r = 0; r < 8; ++r) {
        const int row = rowbase + r;
        s_scores[row][nbase +  0 + l16] = 1.0f / (1.0f + __expf(-acc0[r]));
        s_scores[row][nbase + 16 + l16] = 1.0f / (1.0f + __expf(-acc1[r]));
        s_scores[row][nbase + 32 + l16] = 1.0f / (1.0f + __expf(-acc2[r]));
        s_scores[row][nbase + 48 + l16] = 1.0f / (1.0f + __expf(-acc3[r]));
    }
    __syncthreads();

    // -------------------- group top-2 (8 threads per token) -----------------
    {
        const int token = tid >> 3;      // 0..15
        const int g     = tid & 7;       // 0..7
        float m1 = -3.4e38f, m2 = -3.4e38f;
        const int ebase = g * GRP_SZ;
#pragma unroll 4
        for (int j = 0; j < GRP_SZ; ++j) {
            float v = s_scores[token][ebase + j] + s_bias[ebase + j];
            if (v > m1)      { m2 = m1; m1 = v; }
            else if (v > m2) { m2 = v; }
        }
        s_gscore[token][g] = m1 + m2;
    }
    __syncthreads();

    // -------------------- per-token routing (1 thread per token) ------------
    if (tid < 16) {
        const int token = tid;

        // top-4 groups (stable: ties -> lowest index, like jax.lax.top_k)
        float gs[N_GRP];
#pragma unroll
        for (int g = 0; g < N_GRP; ++g) gs[g] = s_gscore[token][g];
        unsigned gmask = 0;
#pragma unroll
        for (int k = 0; k < TOPK_G; ++k) {
            int   bestg = 0;
            float bestv = gs[0];
#pragma unroll
            for (int g = 1; g < N_GRP; ++g) {
                if (gs[g] > bestv) { bestv = gs[g]; bestg = g; }
            }
            gmask |= (1u << bestg);
            gs[bestg] = -3.4e38f;
        }

        // top-8 experts over masked corrected scores (masked-out -> 0.0f,
        // matching reference's where(score_mask, sfc, 0.0)).
        float bs0 = -3.4e38f, bs1 = -3.4e38f, bs2 = -3.4e38f, bs3 = -3.4e38f;
        float bs4 = -3.4e38f, bs5 = -3.4e38f, bs6 = -3.4e38f, bs7 = -3.4e38f;
        int   bi0 = 0, bi1 = 0, bi2 = 0, bi3 = 0, bi4 = 0, bi5 = 0, bi6 = 0, bi7 = 0;

        for (int e = 0; e < E_DIM; ++e) {
            const bool allowed = (gmask >> (e >> 5)) & 1u;
            float cv = allowed ? (s_scores[token][e] + s_bias[e]) : 0.0f;
            int   ci = e;
            // stable descending insertion (strict > keeps earlier index ahead)
            if (cv > bs0) { float t=bs0; int u=bi0; bs0=cv; bi0=ci; cv=t; ci=u; }
            if (cv > bs1) { float t=bs1; int u=bi1; bs1=cv; bi1=ci; cv=t; ci=u; }
            if (cv > bs2) { float t=bs2; int u=bi2; bs2=cv; bi2=ci; cv=t; ci=u; }
            if (cv > bs3) { float t=bs3; int u=bi3; bs3=cv; bi3=ci; cv=t; ci=u; }
            if (cv > bs4) { float t=bs4; int u=bi4; bs4=cv; bi4=ci; cv=t; ci=u; }
            if (cv > bs5) { float t=bs5; int u=bi5; bs5=cv; bi5=ci; cv=t; ci=u; }
            if (cv > bs6) { float t=bs6; int u=bi6; bs6=cv; bi6=ci; cv=t; ci=u; }
            if (cv > bs7) { bs7=cv; bi7=ci; }
        }

        // gather UNcorrected scores, normalize, scale
        float w0 = s_scores[token][bi0], w1 = s_scores[token][bi1];
        float w2 = s_scores[token][bi2], w3 = s_scores[token][bi3];
        float w4 = s_scores[token][bi4], w5 = s_scores[token][bi5];
        float w6 = s_scores[token][bi6], w7 = s_scores[token][bi7];
        float sum = w0 + w1 + w2 + w3 + w4 + w5 + w6 + w7;
        const float inv = SCALE_R / (sum + 1e-20f);

        const size_t tk = (size_t)(t0 + token);
        float* oi = out + tk * TOPK_E;                       // indices segment
        float* ow = out + (size_t)T * TOPK_E + tk * TOPK_E;  // weights segment
        oi[0] = (float)bi0; oi[1] = (float)bi1; oi[2] = (float)bi2; oi[3] = (float)bi3;
        oi[4] = (float)bi4; oi[5] = (float)bi5; oi[6] = (float)bi6; oi[7] = (float)bi7;
        ow[0] = w0 * inv; ow[1] = w1 * inv; ow[2] = w2 * inv; ow[3] = w3 * inv;
        ow[4] = w4 * inv; ow[5] = w5 * inv; ow[6] = w6 * inv; ow[7] = w7 * inv;
    }
}

extern "C" void kernel_launch(void* const* d_in, const int* in_sizes, int n_in,
                              void* d_out, int out_size, void* d_ws, size_t ws_size,
                              hipStream_t stream) {
    const float* hs   = (const float*)d_in[0];   // [T, 2048] f32
    const float* wt   = (const float*)d_in[1];   // [256, 2048] f32
    const float* bias = (const float*)d_in[2];   // [256] f32
    float* out = (float*)d_out;                  // [T*8 idx | T*8 weights]

    const int T = in_sizes[0] / H_DIM;           // 16384
    dim3 grid(T / 16);
    dim3 block(128);

    const size_t packed_bytes =
        (size_t)(E_DIM / 16) * KBLK * 32 * 16 * sizeof(__bf16);  // 1 MiB

    if (ws_size >= packed_bytes) {
        __bf16* fB = (__bf16*)d_ws;
        const int frag_threads = (E_DIM / 16) * KBLK * 32;       // 32768
        pack_weights_kernel<<<frag_threads / 256, 256, 0, stream>>>(wt, fB);
        topk_router_kernel<true><<<grid, block, 0, stream>>>(hs, wt, fB, bias, out, T);
    } else {
        topk_router_kernel<false><<<grid, block, 0, stream>>>(hs, wt, nullptr, bias, out, T);
    }
}
